// Decoder_70179765617206
// MI455X (gfx1250) — matrix operand order repaired and checked by reading, not compiled, not run on hardware
//
#include <hip/hip_runtime.h>
#include <hip/hip_bf16.h>

// ---------------------------------------------------------------------------
// LSTM captioning decoder for MI455X (gfx1250, wave32).
// 143 GFLOP total, 126 GFLOP in the per-step logits GEMM [64,1024]x[1024,30000].
// W_lin converted to bf16 once per launch (L2-resident: 61MB << 192MB L2);
// hot loops are pure bf16 loads + v_wmma_f32_16x16x32_bf16 (fp32 accumulate).
// Logits GEMM: wave owns an N-tile, iterates 4 M-tiles -> B stream read once,
// 4 independent WMMAs per B tile; A staged into LDS via CDNA5 async-to-LDS
// (global_load_async_to_lds_b128 + s_wait_asynccnt), bank-conflict-free pad.
// ---------------------------------------------------------------------------

typedef __attribute__((ext_vector_type(16))) __bf16 v16bf;
typedef __attribute__((ext_vector_type(8)))  __bf16 v8bf;
typedef __attribute__((ext_vector_type(8)))  float  v8f;
typedef __attribute__((ext_vector_type(4)))  float  v4f;

#define Bsz 64
#define Tseq 32
#define Rreg 49
#define Emb 512
#define Vocab 30000
#define G4E 2048         // 4*Emb
#define K2E 1024         // 2*Emb
#define NTILE (Vocab/16) // 1875
#define APAD 1032        // LDS row stride in bf16 elems (516 dwords; 516%64==4)

// ---- CDNA5 async global->LDS copy (ASYNCcnt path) -------------------------
// vdst = wave-relative LDS byte offset (low 32 bits of a generic &shared ptr,
// per the ISA flat->LDS aperture truncation rule), vaddr = 64-bit global addr.
__device__ __forceinline__ void async_g2l_b128(unsigned int lds_off,
                                               unsigned long long gaddr) {
    asm volatile("global_load_async_to_lds_b128 %0, %1, off"
                 :: "v"(lds_off), "v"(gaddr)
                 : "memory");
}
__device__ __forceinline__ void wait_asynccnt0() {
    asm volatile("s_wait_asynccnt 0" ::: "memory");
}

// ---- WMMA tile loaders (wave32 layouts per CDNA5 ISA 7.12.2) --------------

// A 16x32 bf16 tile, source bf16 row-major [*, ldx].
// M=lane&15, elems 0..15 -> K=half*16..half*16+15 (contiguous, matches B).
__device__ __forceinline__ v16bf load_a_tile(const __bf16* __restrict__ X, int ldx,
                                             int m0, int k0) {
    const int lane = threadIdx.x & 31;
    const int half = lane >> 4;
    const __bf16* row = X + (size_t)(m0 + (lane & 15)) * ldx + k0;
    v8bf c0 = *(const v8bf*)(row + half * 16);
    v8bf c1 = *(const v8bf*)(row + half * 16 + 8);
    v16bf r;
#pragma unroll
    for (int i = 0; i < 8; ++i) { r[i] = c0[i]; r[8 + i] = c1[i]; }
    return r;
}

// B 32x16 tile: B[k][n] = W[n0+n][k0+k]; lane: N=lane&15, K=half*16..+15.
// bf16 source: 16 contiguous bf16 (32B) per lane.
__device__ __forceinline__ v16bf load_b_tile(const __bf16* __restrict__ W, int ldw,
                                             int n0, int k0) {
    const int lane = threadIdx.x & 31;
    const int half = lane >> 4;
    return *(const v16bf*)(W + (size_t)(n0 + (lane & 15)) * ldw + k0 + half * 16);
}

// fp32 source fallback: 16 contiguous floats -> cvt to bf16 in regs.
__device__ __forceinline__ v16bf load_b_tile(const float* __restrict__ W, int ldw,
                                             int n0, int k0) {
    const int lane = threadIdx.x & 31;
    const int half = lane >> 4;
    const v4f* p = (const v4f*)(W + (size_t)(n0 + (lane & 15)) * ldw + k0 + half * 16);
    v4f w0 = p[0], w1 = p[1], w2 = p[2], w3 = p[3];
    v16bf r;
#pragma unroll
    for (int i = 0; i < 4; ++i) {
        r[i]      = (__bf16)w0[i];
        r[4 + i]  = (__bf16)w1[i];
        r[8 + i]  = (__bf16)w2[i];
        r[12 + i] = (__bf16)w3[i];
    }
    return r;
}

// A tile from LDS-staged copy with padded stride APAD (bank-conflict-free:
// each ds_read_b128 touches every bank exactly twice).
__device__ __forceinline__ v16bf load_a_lds(const __bf16* sA, int m0, int k0) {
    const int lane = threadIdx.x & 31;
    const int half = lane >> 4;
    const __bf16* row = sA + (m0 + (lane & 15)) * APAD + k0;
    v8bf c0 = *(const v8bf*)(row + half * 16);
    v8bf c1 = *(const v8bf*)(row + half * 16 + 8);
    v16bf r;
#pragma unroll
    for (int i = 0; i < 8; ++i) { r[i] = c0[i]; r[8 + i] = c1[i]; }
    return r;
}

// ---- utility kernels ------------------------------------------------------

// Bulk fp32 -> bf16 conversion, 8 elems/thread (n must be a multiple of 8).
__global__ void cvt_f32_bf16_kernel(const float* __restrict__ src,
                                    __bf16* __restrict__ dst, int n) {
    const int i = (blockIdx.x * blockDim.x + threadIdx.x) * 8;
    if (i >= n) return;
    v4f a = *(const v4f*)(src + i);
    v4f b = *(const v4f*)(src + i + 4);
    v8bf r;
#pragma unroll
    for (int j = 0; j < 4; ++j) { r[j] = (__bf16)a[j]; r[4 + j] = (__bf16)b[j]; }
    *(v8bf*)(dst + i) = r;
}

// h0 = c0 = mean over R of features, replicated for both layers (+ bf16 shadow).
__global__ void init_state_kernel(const float* __restrict__ features,
                                  float* __restrict__ h, float* __restrict__ c,
                                  __bf16* __restrict__ hb) {
    const int b = blockIdx.x;
    for (int e = threadIdx.x; e < Emb; e += blockDim.x) {
        float s = 0.f;
#pragma unroll 7
        for (int r = 0; r < Rreg; ++r)
            s += features[((size_t)b * Rreg + r) * Emb + e];
        s *= (1.0f / (float)Rreg);
        const int i = b * Emb + e;
        h[i] = s; h[Bsz * Emb + i] = s;
        c[i] = s; c[Bsz * Emb + i] = s;
        hb[i] = (__bf16)s; hb[Bsz * Emb + i] = (__bf16)s;
    }
}

// x0b[b][e] = bf16(embed_table[captions[b][t]][e])
__global__ void embed_gather_kernel(const float* __restrict__ table,
                                    const int* __restrict__ captions,
                                    __bf16* __restrict__ x0b, int t) {
    const int idx = blockIdx.x * blockDim.x + threadIdx.x;   // 0..32767
    const int b = idx >> 9, e = idx & 511;
    const int tok = captions[b * Tseq + t];
    x0b[idx] = (__bf16)table[(size_t)tok * Emb + e];
}

// ---- GEMM kernels ---------------------------------------------------------

// gates[64,2048] = X[64,512] @ Wih^T + H[64,512] @ Whh^T   (bf16 in, f32 acc)
// grid: 128 N-tile blocks x 128 thr (4 waves = 4 M-tiles).
template <typename TB>
__global__ __launch_bounds__(128)
void lstm_gates_kernel(const __bf16* __restrict__ X, const __bf16* __restrict__ H,
                       const TB* __restrict__ Wih, const TB* __restrict__ Whh,
                       float* __restrict__ gates) {
    const int m0 = (threadIdx.x >> 5) * 16;
    const int n0 = blockIdx.x * 16;
    v8f acc = {};
    for (int k0 = 0; k0 < Emb; k0 += 32) {
        v16bf a0 = load_a_tile(X, Emb, m0, k0);
        v16bf b0 = load_b_tile(Wih, Emb, n0, k0);
        acc = __builtin_amdgcn_wmma_f32_16x16x32_bf16(false, a0, false, b0,
                                                      (short)0, acc, false, false);
        v16bf a1 = load_a_tile(H, Emb, m0, k0);
        v16bf b1 = load_b_tile(Whh, Emb, n0, k0);
        acc = __builtin_amdgcn_wmma_f32_16x16x32_bf16(false, a1, false, b1,
                                                      (short)0, acc, false, false);
    }
    const int lane = threadIdx.x & 31;
    const int half = lane >> 4;
    const int n = n0 + (lane & 15);
#pragma unroll
    for (int j = 0; j < 8; ++j)
        gates[(size_t)(m0 + half * 8 + j) * G4E + n] = acc[j];
}

// out[b][t][:] = [h_top|context] @ W_lin^T + b_lin.
// Block = 256 thr (8 waves). Wave g owns N-tile n0=g*16 and all 64 rows:
// 4 accumulators, B stream read once, 4 independent WMMAs per B tile.
// A (64x1024 bf16) staged once per block into LDS via async-to-LDS.
template <typename TB>
__global__ __launch_bounds__(256)
void logits_kernel(const __bf16* __restrict__ Xc,     // [64,1024] bf16
                   const TB* __restrict__ Wlin,       // [30000,1024]
                   const float* __restrict__ blin,
                   float* __restrict__ out, int t) {
    __shared__ __bf16 sA[Bsz * APAD];                  // ~129KB
    {   // async stage: 8192 x 16B chunks, row-aware padded stride (129 uint4/row)
        const unsigned long long src = (unsigned long long)(uintptr_t)Xc;
        const unsigned int dst = (unsigned int)(uintptr_t)&sA[0];
        for (int i = threadIdx.x; i < Bsz * 128; i += 256) {
            const int row = i >> 7, col = i & 127;
            async_g2l_b128(dst + (unsigned int)(row * 129 + col) * 16u,
                           src + (unsigned long long)i * 16ull);
        }
        wait_asynccnt0();
    }
    __syncthreads();
    const int gwave = blockIdx.x * 8 + (threadIdx.x >> 5);
    if (gwave >= NTILE) return;                        // wave-uniform exit
    const int n0 = gwave * 16;
    v8f acc0 = {}, acc1 = {}, acc2 = {}, acc3 = {};
#pragma unroll 4
    for (int k0 = 0; k0 < K2E; k0 += 32) {
        v16bf b = load_b_tile(Wlin, K2E, n0, k0);
        v16bf a0 = load_a_lds(sA,  0, k0);
        v16bf a1 = load_a_lds(sA, 16, k0);
        v16bf a2 = load_a_lds(sA, 32, k0);
        v16bf a3 = load_a_lds(sA, 48, k0);
        acc0 = __builtin_amdgcn_wmma_f32_16x16x32_bf16(false, a0, false, b,
                                                       (short)0, acc0, false, false);
        acc1 = __builtin_amdgcn_wmma_f32_16x16x32_bf16(false, a1, false, b,
                                                       (short)0, acc1, false, false);
        acc2 = __builtin_amdgcn_wmma_f32_16x16x32_bf16(false, a2, false, b,
                                                       (short)0, acc2, false, false);
        acc3 = __builtin_amdgcn_wmma_f32_16x16x32_bf16(false, a3, false, b,
                                                       (short)0, acc3, false, false);
    }
    const int lane = threadIdx.x & 31;
    const int half = lane >> 4;
    const int n = n0 + (lane & 15);
    const float bias = blin[n];
    float* o = out + (size_t)t * Vocab + n;
#pragma unroll
    for (int j = 0; j < 8; ++j) {
        const int m = half * 8 + j;                    // batch row within tile 0
        o[(size_t)(m)      * Tseq * Vocab] = acc0[j] + bias;
        o[(size_t)(m + 16) * Tseq * Vocab] = acc1[j] + bias;
        o[(size_t)(m + 32) * Tseq * Vocab] = acc2[j] + bias;
        o[(size_t)(m + 48) * Tseq * Vocab] = acc3[j] + bias;
    }
}

// ---- pointwise / attention ------------------------------------------------

__device__ __forceinline__ float sigmoidf(float x) {
    return 1.0f / (1.0f + __expf(-x));
}

// LSTM cell update (gate order i,f,g,o); writes fp32 h + bf16 shadow.
__global__ void lstm_cell_kernel(const float* __restrict__ gates,
                                 const float* __restrict__ bih,
                                 const float* __restrict__ bhh,
                                 float* __restrict__ h, float* __restrict__ c,
                                 __bf16* __restrict__ hb) {
    const int idx = blockIdx.x * blockDim.x + threadIdx.x;   // 0..32767
    const int b = idx >> 9, e = idx & 511;
    const float* g = gates + (size_t)b * G4E;
    const float gi = g[e]           + bih[e]           + bhh[e];
    const float gf = g[Emb + e]     + bih[Emb + e]     + bhh[Emb + e];
    const float gg = g[2 * Emb + e] + bih[2 * Emb + e] + bhh[2 * Emb + e];
    const float go = g[3 * Emb + e] + bih[3 * Emb + e] + bhh[3 * Emb + e];
    const float cn = sigmoidf(gf) * c[idx] + sigmoidf(gi) * tanhf(gg);
    const float hn = sigmoidf(go) * tanhf(cn);
    c[idx] = cn;
    h[idx] = hn;
    hb[idx] = (__bf16)hn;
}

// Dot attention + softmax + context; pack [h_top | context] as bf16 rows.
__global__ void attention_kernel(const float* __restrict__ features,
                                 const float* __restrict__ h,      // [2,64,512]
                                 __bf16* __restrict__ xcat) {      // [64,1024]
    __shared__ float sh[Emb];
    __shared__ float salpha[64];
    const int b = blockIdx.x;
    const float* htop = h + (size_t)Bsz * Emb + (size_t)b * Emb;   // layer 1
    for (int e = threadIdx.x; e < Emb; e += blockDim.x) sh[e] = htop[e];
    __syncthreads();
    if (threadIdx.x < Rreg) {
        const float* fr = features + ((size_t)b * Rreg + threadIdx.x) * Emb;
        float s = 0.f;
        for (int e = 0; e < Emb; ++e) s += fr[e] * sh[e];
        salpha[threadIdx.x] = s;
    }
    __syncthreads();
    if (threadIdx.x == 0) {
        float mx = salpha[0];
        for (int r = 1; r < Rreg; ++r) mx = fmaxf(mx, salpha[r]);
        float sum = 0.f;
        for (int r = 0; r < Rreg; ++r) { float ex = __expf(salpha[r] - mx); salpha[r] = ex; sum += ex; }
        const float inv = 1.0f / sum;
        for (int r = 0; r < Rreg; ++r) salpha[r] *= inv;
    }
    __syncthreads();
    for (int e = threadIdx.x; e < Emb; e += blockDim.x) {
        float ctx = 0.f;
        for (int r = 0; r < Rreg; ++r)
            ctx += salpha[r] * features[((size_t)b * Rreg + r) * Emb + e];
        xcat[(size_t)b * K2E + e]       = (__bf16)sh[e];
        xcat[(size_t)b * K2E + Emb + e] = (__bf16)ctx;
    }
}

// ---------------------------------------------------------------------------

extern "C" void kernel_launch(void* const* d_in, const int* in_sizes, int n_in,
                              void* d_out, int out_size, void* d_ws, size_t ws_size,
                              hipStream_t stream) {
    (void)in_sizes; (void)n_in; (void)out_size;

    const float* features = (const float*)d_in[0];   // [64,49,512]
    const int*   captions = (const int*)  d_in[1];   // [64,32]
    const float* table    = (const float*)d_in[2];   // [30000,512]
    const float* W_ih     = (const float*)d_in[3];   // [2,2048,512]
    const float* W_hh     = (const float*)d_in[4];   // [2,2048,512]
    const float* b_ih     = (const float*)d_in[5];   // [2,2048]
    const float* b_hh     = (const float*)d_in[6];   // [2,2048]
    const float* W_lin    = (const float*)d_in[7];   // [30000,1024]
    const float* b_lin    = (const float*)d_in[8];   // [30000]
    float* out = (float*)d_out;                      // [64,32,30000]

    // workspace layout (all chunks 256B-aligned)
    char* ws = (char*)d_ws;
    float*  h     = (float*)(ws + 0);            // 2*64*512*4   = 262144
    float*  c     = (float*)(ws + 262144);       //              = 262144
    float*  gates = (float*)(ws + 524288);       // 64*2048*4    = 524288
    __bf16* xcat  = (__bf16*)(ws + 1048576);     // 64*1024*2    = 131072
    __bf16* x0b   = (__bf16*)(ws + 1179648);     // 64*512*2     = 65536
    __bf16* hb    = (__bf16*)(ws + 1245184);     // 2*64*512*2   = 131072
    __bf16* Wih_b  = (__bf16*)(ws + 1376256);    // 2*2048*512*2 = 4194304
    __bf16* Whh_b  = (__bf16*)(ws + 5570560);    //              = 4194304
    __bf16* Wlin_b = (__bf16*)(ws + 9764864);    // 30000*1024*2 = 61440000
    const size_t FULL_WS = 9764864ull + 61440000ull;     // ~67.9 MB

    const bool fast = (ws_size >= FULL_WS);      // deterministic per-launch

    const size_t Wl = (size_t)G4E * Emb;         // per-layer weight stride
    const size_t HB = (size_t)Bsz * Emb;         // per-layer state stride

    if (fast) {  // one-time bf16 conversion of all GEMM weights (~8us at HBM BW)
        cvt_f32_bf16_kernel<<<(Vocab * K2E) / (8 * 256), 256, 0, stream>>>(W_lin, Wlin_b, Vocab * K2E);
        cvt_f32_bf16_kernel<<<(2 * G4E * Emb) / (8 * 256), 256, 0, stream>>>(W_ih, Wih_b, 2 * G4E * Emb);
        cvt_f32_bf16_kernel<<<(2 * G4E * Emb) / (8 * 256), 256, 0, stream>>>(W_hh, Whh_b, 2 * G4E * Emb);
    }

    init_state_kernel<<<Bsz, 256, 0, stream>>>(features, h, c, hb);

    for (int t = 0; t < Tseq; ++t) {
        embed_gather_kernel<<<128, 256, 0, stream>>>(table, captions, x0b, t);
        if (fast) {
            lstm_gates_kernel<__bf16><<<G4E / 16, 128, 0, stream>>>(
                x0b, hb, Wih_b, Whh_b, gates);
            lstm_cell_kernel<<<128, 256, 0, stream>>>(gates, b_ih, b_hh, h, c, hb);
            lstm_gates_kernel<__bf16><<<G4E / 16, 128, 0, stream>>>(
                hb, hb + HB, Wih_b + Wl, Whh_b + Wl, gates);
            lstm_cell_kernel<<<128, 256, 0, stream>>>(gates, b_ih + G4E, b_hh + G4E,
                                                      h + HB, c + HB, hb + HB);
            attention_kernel<<<Bsz, 256, 0, stream>>>(features, h, xcat);
            logits_kernel<__bf16><<<(NTILE + 7) / 8, 256, 0, stream>>>(
                xcat, Wlin_b, b_lin, out, t);
        } else {  // fallback: cvt fp32 weights in-loop (no big ws needed)
            lstm_gates_kernel<float><<<G4E / 16, 128, 0, stream>>>(
                x0b, hb, W_ih, W_hh, gates);
            lstm_cell_kernel<<<128, 256, 0, stream>>>(gates, b_ih, b_hh, h, c, hb);
            lstm_gates_kernel<float><<<G4E / 16, 128, 0, stream>>>(
                hb, hb + HB, W_ih + Wl, W_hh + Wl, gates);
            lstm_cell_kernel<<<128, 256, 0, stream>>>(gates, b_ih + G4E, b_hh + G4E,
                                                      h + HB, c + HB, hb + HB);
            attention_kernel<<<Bsz, 256, 0, stream>>>(features, h, xcat);
            logits_kernel<float><<<(NTILE + 7) / 8, 256, 0, stream>>>(
                xcat, W_lin, b_lin, out, t);
        }
    }
}
